// HANLayer_90168543412582
// MI455X (gfx1250) — matrix operand order repaired and checked by reading, not compiled, not run on hardware
//
#include <hip/hip_runtime.h>
#include <hip/hip_bf16.h>

// ---------------- types ----------------
typedef __bf16 bf16;
typedef __attribute__((ext_vector_type(8)))  bf16  v8bf;
typedef __attribute__((ext_vector_type(16))) bf16  v16bf;
typedef __attribute__((ext_vector_type(8)))  float v8f;

// ---------------- problem constants ----------------
constexpr int Bsz = 128;
constexpr int S   = 10;
constexpr int D   = 512;
constexpr int DI  = 1024;
constexpr int DSs = 16;
constexpr int DR  = 32;
constexpr int Kc  = 4;      // conv taps
constexpr int L   = 11;
constexpr int Nseq = Bsz * L;        // 1408
constexpr int NL   = Nseq * L;       // 15488 rows through the mamba GEMMs
constexpr int ROWS2 = S * Bsz;       // 1280 rows for FFN / LN

__device__ __forceinline__ float siluf(float x) { return x / (1.f + __expf(-x)); }
__device__ __forceinline__ float softplusf(float x) { return x > 20.f ? x : log1pf(__expf(x)); }
__device__ __forceinline__ int imin(int a, int b) { return a < b ? a : b; }

// ---------------- fp32 -> bf16 convert ----------------
__global__ void f2bf_kernel(const float* __restrict__ in, bf16* __restrict__ out, int n) {
  int i = blockIdx.x * blockDim.x + threadIdx.x;
  if (i < n) out[i] = (bf16)in[i];
}

// ---------------- build fused "am" tensor (B,11,11,D) in bf16 ----------------
__global__ void build_am_kernel(const float* __restrict__ q, const float* __restrict__ v,
                                bf16* __restrict__ am) {
  int idx = blockIdx.x * blockDim.x + threadIdx.x;       // B*11*11*D
  if (idx >= Bsz * L * L * D) return;
  int d = idx & (D - 1);
  int r = idx / D;
  int j = r % L; r /= L;
  int i = r % L;
  int b = r / L;
  float acc = 0.f;
  if (i < S && j < S) acc += q[(b * S + i) * D + d] * v[(b * S + j) * D + d];
  if (j < S)          acc += q[(b * S + j) * D + d];
  if (i < S)          acc += v[(b * S + i) * D + d];
  am[idx] = (bf16)acc;
}

// ---------------- WMMA bf16 GEMM: C[M,N] = A[M,K] @ W[N,K]^T (+bias, act) --------
// Block: 256 threads = 8 wave32s arranged 2(M) x 4(N); macro tile 64x128;
// per-wave 32x32 (2x2 of 16x16 v_wmma_f32_16x16x32_bf16), K-step 32.
// Double-buffered LDS filled by GLOBAL_LOAD_ASYNC_TO_LDS_B128 (ASYNCcnt),
// one s_wait_asynccnt + one barrier per K-step. OOB rows/cols are address-
// clamped (branch-free, EXEC stays all-ones): clamped A-rows / B-cols only
// pollute C entries that are never stored.
// act: 0=none 1=silu 2=relu 3=softplus.  Cf (fp32) / Cb (bf16) optional outputs.
__global__ __launch_bounds__(256)
void gemm_bf16_wmma(const bf16* __restrict__ A, int lda,
                    const bf16* __restrict__ W,
                    const float* __restrict__ bias,
                    float* __restrict__ Cf, bf16* __restrict__ Cb,
                    int M, int N, int Kd, int act) {
  __shared__ __align__(16) bf16 sA[2][64][48];    // stride 96B -> 16B-aligned frag reads
  __shared__ __align__(16) bf16 sB[2][128][48];

  const int tid  = threadIdx.x;
  const int lane = tid & 31;
  const int wave = tid >> 5;        // 0..7
  const int wm   = wave >> 2;       // 0..1
  const int wn   = wave & 3;        // 0..3
  const int m0   = blockIdx.x * 64;
  const int n0   = blockIdx.y * 128;

  const int arow = tid >> 2;            // 0..63
  const int acol = (tid & 3) * 8;       // 0,8,16,24
  const int brow = tid >> 1;            // 0..127
  const int bcol = (tid & 1) * 16;      // 0,16

  // clamped global source pointers (branch-free OOB handling)
  const bf16* pa = A + (size_t)imin(m0 + arow, M - 1) * lda + acol;
  const bf16* pb = W + (size_t)imin(n0 + brow, N - 1) * Kd + bcol;

  // per-thread LDS destinations (low 32 bits of generic pointer == LDS offset)
  unsigned ldsA[2], ldsB[2];
#pragma unroll
  for (int i = 0; i < 2; ++i) {
    ldsA[i] = (unsigned)(size_t)&sA[i][arow][acol];
    ldsB[i] = (unsigned)(size_t)&sB[i][brow][bcol];
  }

  v8f acc[2][2];
#pragma unroll
  for (int r = 0; r < 2; ++r)
#pragma unroll
    for (int c = 0; c < 2; ++c)
#pragma unroll
      for (int e = 0; e < 8; ++e) acc[r][c][e] = 0.f;

  // ISA 7.12.2 16-bit fragment lane offsets
  const int koffA = (lane & 16) ? 8 : 0;    // A: hi-half lanes carry K+8 / K+24
  const int koffB = (lane & 16) ? 8 : 0;    // B: hi-half lanes carry K+8 / K+24
  const int mA0 = wm * 32 + (lane & 15);
  const int nB0 = wn * 32 + (lane & 15);

  union Frag { v16bf v; v8bf h[2]; };

  auto prefetch = [&](int k0, int buf) {
    const bf16* a = pa + k0;     // one 16B chunk of the A tile per thread
    const bf16* b = pb + k0;     // 32B of the B tile per thread (offset applies
                                 // to BOTH lds and global address per ISA 08 §4.4)
    asm volatile("global_load_async_to_lds_b128 %0, %1, off"
                 :: "v"(ldsA[buf]), "v"(a) : "memory");
    asm volatile("global_load_async_to_lds_b128 %0, %1, off"
                 :: "v"(ldsB[buf]), "v"(b) : "memory");
    asm volatile("global_load_async_to_lds_b128 %0, %1, off offset:16"
                 :: "v"(ldsB[buf]), "v"(b) : "memory");
  };

  prefetch(0, 0);
  asm volatile("s_wait_asynccnt 0" ::: "memory");
  __syncthreads();

  int cur = 0;
  for (int k0 = 0; k0 < Kd; k0 += 32) {
    const bool more = (k0 + 32) < Kd;
    if (more) prefetch(k0 + 32, cur ^ 1);   // fill other buffer in background

    Frag af[2], bfr[2];
#pragma unroll
    for (int r = 0; r < 2; ++r) {
      af[r].h[0] = *(const v8bf*)&sA[cur][mA0 + r * 16][koffA];
      af[r].h[1] = *(const v8bf*)&sA[cur][mA0 + r * 16][16 + koffA];
    }
#pragma unroll
    for (int c = 0; c < 2; ++c) {
      bfr[c].h[0] = *(const v8bf*)&sB[cur][nB0 + c * 16][koffB];
      bfr[c].h[1] = *(const v8bf*)&sB[cur][nB0 + c * 16][16 + koffB];
    }
#pragma unroll
    for (int r = 0; r < 2; ++r)
#pragma unroll
      for (int c = 0; c < 2; ++c)
        acc[r][c] = __builtin_amdgcn_wmma_f32_16x16x32_bf16(
            false, af[r].v, false, bfr[c].v, (short)0, acc[r][c], false, false);

    if (more) {
      asm volatile("s_wait_asynccnt 0" ::: "memory");  // next buffer filled
      __syncthreads();                                 // everyone done reading cur
      cur ^= 1;
    }
  }

  // epilogue: C layout (ISA 7.12.2): VGPR e -> M = e (+8 for lanes 16..31), N = lane%16
  const int mhi = (lane & 16) ? 8 : 0;
#pragma unroll
  for (int r = 0; r < 2; ++r)
#pragma unroll
    for (int c = 0; c < 2; ++c)
#pragma unroll
      for (int e = 0; e < 8; ++e) {
        int m = m0 + wm * 32 + r * 16 + mhi + e;
        int n = n0 + wn * 32 + c * 16 + (lane & 15);
        if (m < M && n < N) {
          float vv = acc[r][c][e];
          if (bias) vv += bias[n];
          if (act == 1) vv = siluf(vv);
          else if (act == 2) vv = vv > 0.f ? vv : 0.f;
          else if (act == 3) vv = softplusf(vv);
          size_t o = (size_t)m * N + n;
          if (Cf) Cf[o] = vv;
          if (Cb) Cb[o] = (bf16)vv;
        }
      }
}

// ---------------- causal depthwise conv (K=4) + SiLU ----------------
__global__ void conv_silu_kernel(const bf16* __restrict__ xz,   // NL x 2048, xc = cols [0,DI)
                                 const float* __restrict__ cw,  // DI x 4
                                 const float* __restrict__ cb,
                                 bf16* __restrict__ xconv) {    // NL x DI
  int idx = blockIdx.x * blockDim.x + threadIdx.x;
  if (idx >= NL * DI) return;
  int di = idx & (DI - 1);
  int nt = idx / DI;
  int t  = nt % L;
  int n  = nt / L;
  float acc = cb[di];
#pragma unroll
  for (int k = 0; k < Kc; ++k) {
    int tt = t - (Kc - 1) + k;
    if (tt >= 0) acc += (float)xz[((size_t)(n * L + tt)) * (2 * DI) + di] * cw[di * Kc + k];
  }
  xconv[idx] = (bf16)siluf(acc);
}

// ---------------- selective scan (L=11 recurrence, state in registers) ----------
__global__ __launch_bounds__(1024)
void scan_kernel(const float* __restrict__ dbl,    // NL x 64 (B=32..47, C=48..63)
                 const float* __restrict__ dt,     // NL x DI (post-softplus)
                 const bf16* __restrict__ xconv,   // NL x DI
                 const bf16* __restrict__ xz,      // NL x 2048 (z = cols [DI,2DI))
                 const float* __restrict__ A_log,  // DI x 16
                 const float* __restrict__ D_p,    // DI
                 bf16* __restrict__ y) {           // NL x DI
  int n  = blockIdx.x;     // 0..Nseq-1
  int di = threadIdx.x;    // 0..DI-1
  __shared__ float BC[32];
  float Am[DSs];
#pragma unroll
  for (int s = 0; s < DSs; ++s) Am[s] = -__expf(A_log[di * DSs + s]);
  float h[DSs];
#pragma unroll
  for (int s = 0; s < DSs; ++s) h[s] = 0.f;
  const float Dv = D_p[di];

  for (int t = 0; t < L; ++t) {
    int row = n * L + t;
    if (di < 32) BC[di] = dbl[(size_t)row * 64 + DR + di];
    __syncthreads();
    float dtv = dt[(size_t)row * DI + di];
    float xv  = (float)xconv[(size_t)row * DI + di];
    float dx  = dtv * xv;
    float yv  = 0.f;
#pragma unroll
    for (int s = 0; s < DSs; ++s) {
      float dA = __expf(dtv * Am[s]);
      h[s] = dA * h[s] + dx * BC[s];
      yv += h[s] * BC[DSs + s];
    }
    float zv = (float)xz[(size_t)row * (2 * DI) + DI + di];
    float outv = (yv + Dv * xv) * siluf(zv);
    y[(size_t)row * DI + di] = (bf16)outv;
    __syncthreads();
  }
}

// ---------------- comp: feats_flat[f] = b_op + sum_j out_a_flat[11f+j]*W_op[j] ----
__global__ void comp_kernel(const float* __restrict__ out_a, const float* __restrict__ W_op,
                            const float* __restrict__ b_op, float* __restrict__ feats) {
  int f = blockIdx.x * blockDim.x + threadIdx.x;
  if (f >= Bsz * L * D) return;
  float s = b_op[0];
#pragma unroll
  for (int j = 0; j < L; ++j) s += out_a[(size_t)f * L + j] * W_op[j];
  feats[f] = s;
}

// ---------------- LayerNorm 1: h = LN(src_q^T + src1) ----------------
__global__ __launch_bounds__(256)
void ln1_kernel(const float* __restrict__ src_q, const float* __restrict__ feats,
                const float* __restrict__ g, const float* __restrict__ be,
                float* __restrict__ h, bf16* __restrict__ hb) {
  int row = blockIdx.x;      // t*128 + b
  int t = row >> 7, b = row & 127;
  int tid = threadIdx.x;
  __shared__ float red[256];
  float v[2];
#pragma unroll
  for (int i = 0; i < 2; ++i) {
    int d = tid + i * 256;
    float s1 = 0.5f * (feats[(b * L + t) * D + d] + feats[(b * L + t + 1) * D + d]);
    v[i] = src_q[(b * S + t) * D + d] + s1;
  }
  red[tid] = v[0] + v[1];
  __syncthreads();
  for (int o = 128; o > 0; o >>= 1) { if (tid < o) red[tid] += red[tid + o]; __syncthreads(); }
  float mean = red[0] * (1.f / D);
  __syncthreads();
  red[tid] = (v[0] - mean) * (v[0] - mean) + (v[1] - mean) * (v[1] - mean);
  __syncthreads();
  for (int o = 128; o > 0; o >>= 1) { if (tid < o) red[tid] += red[tid + o]; __syncthreads(); }
  float rs = rsqrtf(red[0] * (1.f / D) + 1e-5f);
#pragma unroll
  for (int i = 0; i < 2; ++i) {
    int d = tid + i * 256;
    float o = (v[i] - mean) * rs * g[d] + be[d];
    h[(size_t)row * D + d] = o;
    hb[(size_t)row * D + d] = (bf16)o;
  }
}

// ---------------- LayerNorm 2 + output transpose -> (B, S, D) ----------------
__global__ __launch_bounds__(256)
void ln2_kernel(const float* __restrict__ h, const float* __restrict__ ff,
                const float* __restrict__ g, const float* __restrict__ be,
                float* __restrict__ out) {
  int row = blockIdx.x;      // t*128 + b
  int t = row >> 7, b = row & 127;
  int tid = threadIdx.x;
  __shared__ float red[256];
  float v[2];
#pragma unroll
  for (int i = 0; i < 2; ++i) {
    int d = tid + i * 256;
    v[i] = h[(size_t)row * D + d] + ff[(size_t)row * D + d];
  }
  red[tid] = v[0] + v[1];
  __syncthreads();
  for (int o = 128; o > 0; o >>= 1) { if (tid < o) red[tid] += red[tid + o]; __syncthreads(); }
  float mean = red[0] * (1.f / D);
  __syncthreads();
  red[tid] = (v[0] - mean) * (v[0] - mean) + (v[1] - mean) * (v[1] - mean);
  __syncthreads();
  for (int o = 128; o > 0; o >>= 1) { if (tid < o) red[tid] += red[tid + o]; __syncthreads(); }
  float rs = rsqrtf(red[0] * (1.f / D) + 1e-5f);
#pragma unroll
  for (int i = 0; i < 2; ++i) {
    int d = tid + i * 256;
    out[((size_t)b * S + t) * D + d] = (v[i] - mean) * rs * g[d] + be[d];
  }
}

// ---------------- host ----------------
extern "C" void kernel_launch(void* const* d_in, const int* in_sizes, int n_in,
                              void* d_out, int out_size, void* d_ws, size_t ws_size,
                              hipStream_t stream) {
  (void)in_sizes; (void)n_in; (void)out_size; (void)ws_size;
  const float* src_q = (const float*)d_in[0];
  const float* src_v = (const float*)d_in[1];
  const float* W_in  = (const float*)d_in[2];
  const float* convw = (const float*)d_in[3];
  const float* convb = (const float*)d_in[4];
  const float* W_x   = (const float*)d_in[5];
  const float* W_dt  = (const float*)d_in[6];
  const float* b_dt  = (const float*)d_in[7];
  const float* A_log = (const float*)d_in[8];
  const float* D_p   = (const float*)d_in[9];
  const float* W_out = (const float*)d_in[10];
  const float* W_op  = (const float*)d_in[11];
  const float* b_op  = (const float*)d_in[12];
  const float* W1    = (const float*)d_in[13];
  const float* b1    = (const float*)d_in[14];
  const float* W2    = (const float*)d_in[15];
  const float* b2    = (const float*)d_in[16];
  const float* g1    = (const float*)d_in[17];
  const float* be1   = (const float*)d_in[18];
  const float* g2    = (const float*)d_in[19];
  const float* be2   = (const float*)d_in[20];
  float* outp = (float*)d_out;

  // --- carve workspace ---
  char* p = (char*)d_ws;
  auto take = [&](size_t bytes) { char* r = p; p += (bytes + 255) & ~(size_t)255; return r; };
  bf16* wIn_b   = (bf16*)take((size_t)2 * DI * D * 2);
  bf16* wX_b    = (bf16*)take((size_t)(DR + 2 * DSs) * DI * 2);
  bf16* wDt_b   = (bf16*)take((size_t)DI * DR * 2);
  bf16* wOut_b  = (bf16*)take((size_t)D * DI * 2);
  bf16* w1_b    = (bf16*)take((size_t)D * D * 2);
  bf16* w2_b    = (bf16*)take((size_t)D * D * 2);
  bf16* am_b    = (bf16*)take((size_t)NL * D * 2);
  bf16* xz_b    = (bf16*)take((size_t)NL * 2 * DI * 2);
  bf16* xconv_b = (bf16*)take((size_t)NL * DI * 2);
  float* dbl_f  = (float*)take((size_t)NL * 64 * 4);
  bf16* dbl_b   = (bf16*)take((size_t)NL * 64 * 2);
  float* dt_f   = (float*)take((size_t)NL * DI * 4);
  bf16* y_b     = (bf16*)take((size_t)NL * DI * 2);
  float* outa_f = (float*)take((size_t)NL * D * 4);
  float* feats_f= (float*)take((size_t)Bsz * L * D * 4);
  float* h_f    = (float*)take((size_t)ROWS2 * D * 4);
  bf16* h_b     = (bf16*)take((size_t)ROWS2 * D * 2);
  bf16* ff1_b   = (bf16*)take((size_t)ROWS2 * D * 2);
  float* ff2_f  = (float*)take((size_t)ROWS2 * D * 4);

  auto cdiv = [](int a, int b) { return (a + b - 1) / b; };

  // weights -> bf16
  f2bf_kernel<<<cdiv(2 * DI * D, 256), 256, 0, stream>>>(W_in, wIn_b, 2 * DI * D);
  f2bf_kernel<<<cdiv(64 * DI, 256), 256, 0, stream>>>(W_x, wX_b, 64 * DI);
  f2bf_kernel<<<cdiv(DI * DR, 256), 256, 0, stream>>>(W_dt, wDt_b, DI * DR);
  f2bf_kernel<<<cdiv(D * DI, 256), 256, 0, stream>>>(W_out, wOut_b, D * DI);
  f2bf_kernel<<<cdiv(D * D, 256), 256, 0, stream>>>(W1, w1_b, D * D);
  f2bf_kernel<<<cdiv(D * D, 256), 256, 0, stream>>>(W2, w2_b, D * D);

  // fused tensor
  build_am_kernel<<<cdiv(Bsz * L * L * D, 256), 256, 0, stream>>>(src_q, src_v, am_b);

  // xz = am @ W_in.T              (15488 x 2048, K=512)
  gemm_bf16_wmma<<<dim3(NL / 64, cdiv(2 * DI, 128)), 256, 0, stream>>>(
      am_b, D, wIn_b, nullptr, nullptr, xz_b, NL, 2 * DI, D, 0);

  // causal conv + silu
  conv_silu_kernel<<<cdiv(NL * DI, 256), 256, 0, stream>>>(xz_b, convw, convb, xconv_b);

  // dbl = xconv @ W_x.T           (15488 x 64, K=1024)
  gemm_bf16_wmma<<<dim3(NL / 64, 1), 256, 0, stream>>>(
      xconv_b, DI, wX_b, nullptr, dbl_f, dbl_b, NL, 64, DI, 0);

  // dt = softplus(dbl[:, :32] @ W_dt.T + b_dt)   (15488 x 1024, K=32)
  gemm_bf16_wmma<<<dim3(NL / 64, cdiv(DI, 128)), 256, 0, stream>>>(
      dbl_b, 64, wDt_b, b_dt, dt_f, nullptr, NL, DI, DR, 3);

  // selective scan + D-skip + z-gate
  scan_kernel<<<Nseq, DI, 0, stream>>>(dbl_f, dt_f, xconv_b, xz_b, A_log, D_p, y_b);

  // out_a = y @ W_out.T           (15488 x 512, K=1024)
  gemm_bf16_wmma<<<dim3(NL / 64, cdiv(D, 128)), 256, 0, stream>>>(
      y_b, DI, wOut_b, nullptr, outa_f, nullptr, NL, D, DI, 0);

  // strided 11-tap projection -> feats
  comp_kernel<<<cdiv(Bsz * L * D, 256), 256, 0, stream>>>(outa_f, W_op, b_op, feats_f);

  // LN1
  ln1_kernel<<<ROWS2, 256, 0, stream>>>(src_q, feats_f, g1, be1, h_f, h_b);

  // FFN: ff1 = relu(h @ W1.T + b1); ff2 = ff1 @ W2.T + b2
  gemm_bf16_wmma<<<dim3(ROWS2 / 64, cdiv(D, 128)), 256, 0, stream>>>(
      h_b, D, w1_b, b1, nullptr, ff1_b, ROWS2, D, D, 2);
  gemm_bf16_wmma<<<dim3(ROWS2 / 64, cdiv(D, 128)), 256, 0, stream>>>(
      ff1_b, D, w2_b, b2, ff2_f, nullptr, ROWS2, D, D, 0);

  // LN2 + output transpose
  ln2_kernel<<<ROWS2, 256, 0, stream>>>(h_f, ff2_f, g2, be2, outp);
}
